// MultiHeadHierarchicalAttention_19688130085319
// MI455X (gfx1250) — compile-verified, matrix operand order checked by reading, not run on hardware
//
#include <hip/hip_runtime.h>

// MultiHeadHierarchicalAttention for MI455X (gfx1250, wave32, WMMA f16->f32).
// Pipeline: 3x proj GEMM (f32->f16 WMMA, head-permuted f16 out, Q pre-scaled by
//           log2e/sqrt(dk)) -> flash attention in base-2 domain (WMMA QK^T,
//           online softmax with WMMA-computed row sums, WMMA PV)
//           -> output GEMM (f16 WMMA, f32 out).
// The scalar logit bias (BIAS_S + GAUSS_W = -0.4) is softmax-invariant: dropped.

typedef _Float16 v16h __attribute__((ext_vector_type(16)));
typedef _Float16 v8h  __attribute__((ext_vector_type(8)));
typedef _Float16 v4h  __attribute__((ext_vector_type(4)));
typedef __fp16   v2fp __attribute__((ext_vector_type(2)));   // cvt_pkrtz result type
typedef float    v8f  __attribute__((ext_vector_type(8)));

static __device__ __forceinline__ v8f wmma16(v16h a, v16h b, v8f c) {
    return __builtin_amdgcn_wmma_f32_16x16x32_f16(false, a, false, b, (short)0, c, false, false);
}
static __device__ __forceinline__ v16h pack16(v8h lo, v8h hi) {
    v16h r;
#pragma unroll
    for (int i = 0; i < 8; ++i) { r[i] = lo[i]; r[i + 8] = hi[i]; }
    return r;
}
static __device__ __forceinline__ v8f zero8() {
    v8f z;
#pragma unroll
    for (int i = 0; i < 8; ++i) z[i] = 0.0f;
    return z;
}
// Raw v_exp_f32 (base-2): no denormal-range guard sequence; underflow -> 0,
// which is exactly what softmax weights need.
static __device__ __forceinline__ float fast_exp2(float x) {
    return __builtin_amdgcn_exp2f(x);
}
// float4 -> 4 halfs via v_cvt_pk_rtz_f16_f32 (one packed convert per pair)
static __device__ __forceinline__ v4h cvt4(float4 v) {
    v2fp lo = __builtin_amdgcn_cvt_pkrtz(v.x, v.y);
    v2fp hi = __builtin_amdgcn_cvt_pkrtz(v.z, v.w);
    v4h r;
    r[0] = (_Float16)lo[0]; r[1] = (_Float16)lo[1];
    r[2] = (_Float16)hi[0]; r[3] = (_Float16)hi[1];
    return r;
}

// ---------------------------------------------------------------------------
// C = A(MxK) @ W(NxK)^T + bias, N fixed 1024, K fixed 1024 at call sites.
// MODE 0: out f16, permuted to [b, h, l, d] (head split), times `scale`.
// MODE 1: out f32, row-major [M, 1024].
// ---------------------------------------------------------------------------
template <typename AT, int MODE>
__global__ __launch_bounds__(256) void gemm_xWt_kernel(
    const AT* __restrict__ A, const float* __restrict__ W,
    const float* __restrict__ bias, void* __restrict__ outp,
    float scale, int M, int K, int L)
{
    __shared__ _Float16 sA[128][32];
    __shared__ _Float16 sB[128][32];
    const int tid = threadIdx.x;
    const int wave = tid >> 5, lane = tid & 31;
    const int lm = lane & 15, lh = lane >> 4;
    const int m0 = blockIdx.y * 128, n0 = blockIdx.x * 128;
    const int wm = (wave & 3) * 32, wn = (wave >> 2) * 64;

    v8f acc[2][4];
#pragma unroll
    for (int mi = 0; mi < 2; ++mi)
#pragma unroll
        for (int ni = 0; ni < 4; ++ni) acc[mi][ni] = zero8();

    for (int k0 = 0; k0 < K; k0 += 32) {
        // ---- stage A tile (convert to f16 in LDS, packed converts) ----
        if constexpr (sizeof(AT) == 4) {
#pragma unroll
            for (int i = tid; i < 1024; i += 256) {
                int row = i >> 3, c4 = (i & 7) * 4;
                float4 v = *(const float4*)((const float*)A + (size_t)(m0 + row) * K + k0 + c4);
                *(v4h*)(&sA[row][c4]) = cvt4(v);
            }
        } else {
#pragma unroll
            for (int i = tid; i < 512; i += 256) {
                int row = i >> 2, c8 = (i & 3) * 8;
                *(v8h*)(&sA[row][c8]) =
                    *(const v8h*)((const _Float16*)A + (size_t)(m0 + row) * K + k0 + c8);
            }
        }
        // ---- stage W tile (f32 -> f16, packed converts) ----
#pragma unroll
        for (int i = tid; i < 1024; i += 256) {
            int row = i >> 3, c4 = (i & 7) * 4;
            float4 v = *(const float4*)(W + (size_t)(n0 + row) * K + k0 + c4);
            *(v4h*)(&sB[row][c4]) = cvt4(v);
        }
        __syncthreads();

        v16h a[2];
#pragma unroll
        for (int mi = 0; mi < 2; ++mi) {
            int row = wm + mi * 16 + lm;
            a[mi] = pack16(*(const v8h*)(&sA[row][lh * 8]),
                           *(const v8h*)(&sA[row][16 + lh * 8]));
        }
#pragma unroll
        for (int ni = 0; ni < 4; ++ni) {
            int row = wn + ni * 16 + lm;
            v16h b = pack16(*(const v8h*)(&sB[row][lh * 8]),
                            *(const v8h*)(&sB[row][16 + lh * 8]));
#pragma unroll
            for (int mi = 0; mi < 2; ++mi)
                acc[mi][ni] = wmma16(a[mi], b, acc[mi][ni]);
        }
        __syncthreads();
    }

    // ---- epilogue ----
#pragma unroll
    for (int mi = 0; mi < 2; ++mi)
#pragma unroll
        for (int ni = 0; ni < 4; ++ni)
#pragma unroll
            for (int r = 0; r < 8; ++r) {
                int grow = m0 + wm + mi * 16 + r + lh * 8;   // C layout: M = r + 8*(lane>=16)
                int gcol = n0 + wn + ni * 16 + lm;           //           N = lane % 16
                float val = acc[mi][ni][r] + bias[gcol];
                if constexpr (MODE == 0) {
                    int b = grow / L, l = grow % L;
                    int h = gcol >> 6, d = gcol & 63;
                    ((_Float16*)outp)[(((size_t)b * 16 + h) * L + l) * 64 + d] =
                        (_Float16)(val * scale);
                } else {
                    ((float*)outp)[(size_t)grow * 1024 + gcol] = val;
                }
            }
}

// ---------------------------------------------------------------------------
// Flash attention per (b, h). Block = 256 threads = 8 waves; each block owns
// 128 q-rows (16 per wave). Q carries log2e/sqrt(dk); softmax runs in base 2.
// Row sums of P computed on the matrix unit: rs = P @ ones (C-layout result
// puts row r's sum straight into vgpr r of every lane in the half-group).
// ---------------------------------------------------------------------------
__global__ __launch_bounds__(256) void attn_kernel(
    const _Float16* __restrict__ Q, const _Float16* __restrict__ K,
    const _Float16* __restrict__ V, _Float16* __restrict__ Ctx, int L)
{
    const int H = 16;
    const int qt = blockIdx.x, h = blockIdx.y, b = blockIdx.z;
    const int tid = threadIdx.x;
    const int wave = tid >> 5, lane = tid & 31;
    const int lm = lane & 15, lh = lane >> 4;

    const size_t bh = ((size_t)b * H + h) * (size_t)L * 64;
    const _Float16* Qp = Q + bh;   // [L][64]
    const _Float16* Kp = K + bh;
    const _Float16* Vp = V + bh;

    __shared__ _Float16 sK[64][64];       // [key][d]    (natural)
    __shared__ _Float16 sV[64][64];       // [d][key]    (transposed)
    __shared__ _Float16 sP[8][16][64];    // per-wave P tile [qrow][key]

    const int qrow0 = qt * 128 + wave * 16;

    // Resident Q fragments (2 k-chunks of 32 over d=64)
    v16h qfrag[2];
    {
        const _Float16* qr = Qp + (size_t)(qrow0 + lm) * 64;
#pragma unroll
        for (int kk = 0; kk < 2; ++kk)
            qfrag[kk] = pack16(*(const v8h*)(qr + kk * 32 + lh * 8),
                               *(const v8h*)(qr + kk * 32 + 16 + lh * 8));
    }
    v16h ones;
#pragma unroll
    for (int i = 0; i < 16; ++i) ones[i] = (_Float16)1.0f;

    float m_r[8], l_r[8];
    v8f o[4];
#pragma unroll
    for (int r = 0; r < 8; ++r) { m_r[r] = -1e30f; l_r[r] = 0.0f; }
#pragma unroll
    for (int di = 0; di < 4; ++di) o[di] = zero8();

    for (int k0 = 0; k0 < L; k0 += 64) {
        // ---- cooperative K/V tile load (V transposed) ----
#pragma unroll
        for (int i = tid; i < 512; i += 256) {
            int key = i >> 3, c8 = (i & 7) * 8;
            *(v8h*)(&sK[key][c8]) = *(const v8h*)(Kp + (size_t)(k0 + key) * 64 + c8);
            v8h vd = *(const v8h*)(Vp + (size_t)(k0 + key) * 64 + c8);
#pragma unroll
            for (int j = 0; j < 8; ++j) sV[c8 + j][key] = vd[j];
        }
        __syncthreads();

        // ---- S = Q @ K^T : 16 x 64 base-2 logits per wave ----
        v8f s[4];
#pragma unroll
        for (int ni = 0; ni < 4; ++ni) {
            v8f a = zero8();
            int krow = ni * 16 + lm;
#pragma unroll
            for (int kk = 0; kk < 2; ++kk) {
                v16h bf = pack16(*(const v8h*)(&sK[krow][kk * 32 + lh * 8]),
                                 *(const v8h*)(&sK[krow][kk * 32 + 16 + lh * 8]));
                a = wmma16(qfrag[kk], bf, a);
            }
            s[ni] = a;
        }

        // ---- online softmax: cross-lane row max, raw v_exp_f32 in C-layout ----
        float sc_r[8];
#pragma unroll
        for (int r = 0; r < 8; ++r) {
            float mx = -1e30f;
#pragma unroll
            for (int ni = 0; ni < 4; ++ni) mx = fmaxf(mx, s[ni][r]);
#pragma unroll
            for (int off = 1; off < 16; off <<= 1) mx = fmaxf(mx, __shfl_xor(mx, off, 32));
            float mnew = fmaxf(m_r[r], mx);
            sc_r[r] = fast_exp2(m_r[r] - mnew);
            m_r[r] = mnew;
#pragma unroll
            for (int ni = 0; ni < 4; ++ni) s[ni][r] = fast_exp2(s[ni][r] - mnew);
        }
#pragma unroll
        for (int di = 0; di < 4; ++di)
#pragma unroll
            for (int r = 0; r < 8; ++r) o[di][r] = o[di][r] * sc_r[r];

        // ---- P: C-layout -> A-layout via per-wave LDS tile ----
#pragma unroll
        for (int ni = 0; ni < 4; ++ni)
#pragma unroll
            for (int r = 0; r < 8; ++r)
                sP[wave][r + lh * 8][ni * 16 + lm] = (_Float16)s[ni][r];
        asm volatile("s_wait_dscnt 0" ::: "memory");  // own-wave LDS RAW

        v16h pfrag[2];
#pragma unroll
        for (int kk = 0; kk < 2; ++kk) {
            const _Float16* pr = &sP[wave][lm][kk * 32];
            pfrag[kk] = pack16(*(const v8h*)(pr + lh * 8),
                               *(const v8h*)(pr + 16 + lh * 8));
        }

        // ---- row sums on the matrix unit: rs = P @ ones ----
        v8f rs = wmma16(pfrag[0], ones, zero8());
        rs = wmma16(pfrag[1], ones, rs);
#pragma unroll
        for (int r = 0; r < 8; ++r) l_r[r] = l_r[r] * sc_r[r] + rs[r];

        // ---- O += P @ V ----
#pragma unroll
        for (int kk = 0; kk < 2; ++kk)
#pragma unroll
            for (int di = 0; di < 4; ++di) {
                int drow = di * 16 + lm;
                v16h vf = pack16(*(const v8h*)(&sV[drow][kk * 32 + lh * 8]),
                                 *(const v8h*)(&sV[drow][kk * 32 + 16 + lh * 8]));
                o[di] = wmma16(pfrag[kk], vf, o[di]);
            }
        __syncthreads();
    }

    // ---- normalize and store ctx as [b, l, h*64 + d] f16 ----
#pragma unroll
    for (int r = 0; r < 8; ++r) {
        float inv = 1.0f / l_r[r];
        int row = qrow0 + r + lh * 8;
        size_t base = ((size_t)b * L + row) * 1024 + (size_t)h * 64;
#pragma unroll
        for (int di = 0; di < 4; ++di)
            Ctx[base + di * 16 + lm] = (_Float16)(o[di][r] * inv);
    }
}

extern "C" void kernel_launch(void* const* d_in, const int* in_sizes, int n_in,
                              void* d_out, int out_size, void* d_ws, size_t ws_size,
                              hipStream_t stream) {
    (void)in_sizes; (void)n_in; (void)out_size; (void)ws_size;
    const float* q    = (const float*)d_in[0];
    // d_in[1], d_in[2] (k_w, v_w) are unused by the reference
    const float* k_s  = (const float*)d_in[3];
    const float* v_s  = (const float*)d_in[4];
    const float* wq   = (const float*)d_in[5];
    const float* bq   = (const float*)d_in[6];
    const float* wk   = (const float*)d_in[7];
    const float* bk   = (const float*)d_in[8];
    const float* wv   = (const float*)d_in[9];
    const float* bv   = (const float*)d_in[10];
    const float* wfc  = (const float*)d_in[11];
    const float* bfc  = (const float*)d_in[12];

    const int B = 4, L = 2048, D = 1024, H = 16;
    const int M = B * L;                        // 8192
    const size_t per = (size_t)B * H * L * 64;  // 8M halves each

    _Float16* Qh = (_Float16*)d_ws;
    _Float16* Kh = Qh + per;
    _Float16* Vh = Kh + per;
    _Float16* Ct = Vh + per;

    dim3 gg(D / 128, M / 128);  // 8 x 64 blocks

    // Q scale = log2(e)/sqrt(64): softmax runs in base-2 (v_exp_f32 native).
    const float qscale = 0.125f * 1.44269504088896340736f;

    gemm_xWt_kernel<float, 0><<<gg, 256, 0, stream>>>(q,   wq, bq, Qh, qscale, M, D, L);
    gemm_xWt_kernel<float, 0><<<gg, 256, 0, stream>>>(k_s, wk, bk, Kh, 1.0f,   M, D, L);
    gemm_xWt_kernel<float, 0><<<gg, 256, 0, stream>>>(v_s, wv, bv, Vh, 1.0f,   M, D, L);

    attn_kernel<<<dim3(L / 128, H, B), 256, 0, stream>>>(Qh, Kh, Vh, Ct, L);

    gemm_xWt_kernel<_Float16, 1><<<gg, 256, 0, stream>>>(Ct, wfc, bfc, d_out, 1.0f, M, D, L);
}
